// M_EfficientSelfAtten_3_68375879352568
// MI455X (gfx1250) — matrix operand order repaired and checked by reading, not compiled, not run on hardware
//
#include <hip/hip_runtime.h>
#include <hip/hip_bf16.h>
#include <cstdint>

// ---------------------------------------------------------------------------
// Shapes: B=16, N=5880, C=64, M=735 (padded to 736), Npad=5888 (92 x 64)
// Workspace layout (bytes), total ~18.1 MB:
//   Q   f16 [16][5888][64]      @ 0          (12,058,624)
//   X1  f32 [16][736][64]       @ 12,058,624 ( 3,014,656)
//   K   f16 [16][736][64]       @ 15,073,280 ( 1,507,328)
//   Vt  f16 [16][64][736]       @ 16,580,608 ( 1,507,328)   (contiguous after K!)
// ---------------------------------------------------------------------------

typedef _Float16 v16h __attribute__((ext_vector_type(16)));
typedef float    v8f  __attribute__((ext_vector_type(8)));
typedef unsigned int uint32x4 __attribute__((ext_vector_type(4)));
typedef int          int32x8  __attribute__((ext_vector_type(8)));
typedef int          int32x4  __attribute__((ext_vector_type(4)));

union AFrag { v16h v; uint32_t u[8]; };
union DFrag { v8f  v; float    f[8]; };

#define NB 16
#define NQ 5880
#define NQPAD 5888
#define CD 64
#define MK 735
#define MKPAD 736

// 16-lane all-reduce max step via DPP (pure VALU, no LDS/bpermute traffic).
template <int CTRL>
__device__ __forceinline__ float dpp_fmax_step(float x) {
  int y = __builtin_amdgcn_update_dpp(0, __float_as_int(x), CTRL, 0xF, 0xF, true);
  return fmaxf(x, __int_as_float(y));
}

// ---------------- Kernel 1: q = (x*Wq + bq) * 1/sqrt(C), f16 ----------------
__global__ __launch_bounds__(256) void qproj_kernel(
    const float* __restrict__ x, const float* __restrict__ Wq,
    const float* __restrict__ bq, _Float16* __restrict__ Q) {
  __shared__ float Ws[64 * 64];
  __shared__ float Xs[64 * 64];
  const int b = blockIdx.x / 92, tile = blockIdx.x % 92;
  for (int i = threadIdx.x; i < 4096; i += 256) {
    Ws[i] = Wq[i];
    int r = i >> 6, c = i & 63;
    int n = tile * 64 + r;
    Xs[i] = (n < NQ) ? x[((size_t)b * NQ + n) * 64 + c] : 0.f;
  }
  __syncthreads();
  const int r = threadIdx.x >> 2;
  const int cg = (threadIdx.x & 3) * 16;
  const int n = tile * 64 + r;
  if (n < NQ) {
    float o[16];
#pragma unroll
    for (int i = 0; i < 16; ++i) o[i] = bq[cg + i];
    for (int k = 0; k < 64; ++k) {
      float xv = Xs[r * 64 + k];
#pragma unroll
      for (int i = 0; i < 16; ++i) o[i] += xv * Ws[k * 64 + cg + i];
    }
    _Float16* qo = Q + ((size_t)b * NQPAD + n) * 64 + cg;
#pragma unroll
    for (int i = 0; i < 16; ++i) qo[i] = (_Float16)(o[i] * 0.125f);
  }
}

// ---------------- Kernel 2: copy tem0 rows + zero pad row 735 ---------------
__global__ __launch_bounds__(256) void copy_zero_kernel(
    const float* __restrict__ x, float* __restrict__ X1) {
  int idx = blockIdx.x * 256 + threadIdx.x;  // 16*393*64
  int c = idx & 63, rest = idx >> 6;
  int m = rest % 393, b = rest / 393;
  if (m < 392)
    X1[((size_t)b * MKPAD + m) * 64 + c] = x[((size_t)b * NQ + m) * 64 + c];
  else
    X1[((size_t)b * MKPAD + MK) * 64 + c] = 0.f;
}

// ---------------- Kernel 3: sr1 8x8/s8 conv (rows 392..440) -----------------
__global__ __launch_bounds__(256) void conv1_kernel(
    const float* __restrict__ x, const float* __restrict__ w,
    const float* __restrict__ bias, float* __restrict__ X1) {
  int idx = blockIdx.x * 256 + threadIdx.x;  // 16*49*64
  int oc = idx & 63, rest = idx >> 6;
  int p = rest % 49, b = rest / 49;
  int ph = p / 7, pw = p % 7;
  float acc = bias[oc];
  for (int kh = 0; kh < 8; ++kh)
    for (int kw = 0; kw < 8; ++kw) {
      const float* xr = x + ((size_t)b * NQ + 392 + (ph * 8 + kh) * 56 + pw * 8 + kw) * 64;
      const float* wr = w + ((size_t)oc * 64) * 64 + kh * 8 + kw;
#pragma unroll 8
      for (int ic = 0; ic < 64; ++ic) acc += xr[ic] * wr[ic * 64];
    }
  X1[((size_t)b * MKPAD + 392 + p) * 64 + oc] = acc;
}

// ---------------- Kernel 4: sr3 4x4/s4 conv (rows 441..538) -----------------
__global__ __launch_bounds__(256) void conv3_kernel(
    const float* __restrict__ x, const float* __restrict__ w,
    const float* __restrict__ bias, float* __restrict__ X1) {
  int idx = blockIdx.x * 256 + threadIdx.x;  // 16*49*128
  int oc = idx & 127, rest = idx >> 7;
  int s = rest % 49, b = rest / 49;
  int ph = s / 7, pw = s % 7;
  float acc = bias[oc];
  for (int kh = 0; kh < 4; ++kh)
    for (int kw = 0; kw < 4; ++kw) {
      const float* xr = x + ((size_t)b * NQ + 3528 + (4 * ph + kh) * 56 + 2 * (4 * pw + kw)) * 64;
      const float* wr = w + ((size_t)oc * 128) * 16 + kh * 4 + kw;
#pragma unroll 8
      for (int cc = 0; cc < 128; ++cc) acc += xr[cc] * wr[cc * 16];
    }
  int i = oc >> 1, j = (oc & 1) * 49 + s;
  X1[((size_t)b * MKPAD + 441 + j) * 64 + i] = acc;
}

// ---------------- Kernel 5: sr5 2x2/s2 conv (rows 539..734) -----------------
__global__ __launch_bounds__(256) void conv5_kernel(
    const float* __restrict__ x, const float* __restrict__ w,
    const float* __restrict__ bias, float* __restrict__ X1) {
  int idx = blockIdx.x * 256 + threadIdx.x;  // 16*49*256
  int oc = idx & 255, rest = idx >> 8;
  int s = rest % 49, b = rest / 49;
  int ph = s / 7, pw = s % 7;
  float acc = bias[oc];
  for (int kh = 0; kh < 2; ++kh)
    for (int kw = 0; kw < 2; ++kw) {
      const float* xr = x + ((size_t)b * NQ + 5096 + (2 * ph + kh) * 56 + 4 * (2 * pw + kw)) * 64;
      const float* wr = w + ((size_t)oc * 256) * 4 + kh * 2 + kw;
#pragma unroll 8
      for (int cc = 0; cc < 256; ++cc) acc += xr[cc] * wr[cc * 4];
    }
  int i = oc >> 2, j = (oc & 3) * 49 + s;
  X1[((size_t)b * MKPAD + 539 + j) * 64 + i] = acc;
}

// -------- Kernel 6: LayerNorm + exact GELU + kv GEMM -> K f16, Vt f16 -------
__global__ __launch_bounds__(256) void lngelu_kv_kernel(
    const float* __restrict__ X1, const float* __restrict__ kvW,
    const float* __restrict__ kvB, const float* __restrict__ lng,
    const float* __restrict__ lnb, _Float16* __restrict__ K,
    _Float16* __restrict__ Vt) {
  __shared__ float wS[64 * 128];
  __shared__ float bS[128];
  for (int i = threadIdx.x; i < 64 * 128; i += 256) wS[i] = kvW[i];
  if (threadIdx.x < 128) bS[threadIdx.x] = kvB[threadIdx.x];
  __syncthreads();
  const int lane = threadIdx.x & 31, wid = threadIdx.x >> 5;
  const int row = blockIdx.x * 8 + wid;  // 0..11775
  const int b = row / MKPAD, m = row % MKPAD;
  const bool pad = (m == MK);
  float g0 = 0.f, g1 = 0.f;
  if (!pad) {
    const float* xr = X1 + ((size_t)b * MKPAD + m) * 64;
    float v0 = xr[lane], v1 = xr[lane + 32];
    float s = v0 + v1, s2 = v0 * v0 + v1 * v1;
#pragma unroll
    for (int d = 1; d < 32; d <<= 1) {
      s += __shfl_xor(s, d, 32);
      s2 += __shfl_xor(s2, d, 32);
    }
    float mu = s * (1.f / 64.f);
    float var = s2 * (1.f / 64.f) - mu * mu;
    float rstd = rsqrtf(var + 1e-5f);
    g0 = (v0 - mu) * rstd * lng[lane] + lnb[lane];
    g1 = (v1 - mu) * rstd * lng[lane + 32] + lnb[lane + 32];
    g0 = 0.5f * g0 * (1.f + erff(g0 * 0.70710678f));
    g1 = 0.5f * g1 * (1.f + erff(g1 * 0.70710678f));
  }
  float acc[4] = {0.f, 0.f, 0.f, 0.f};
  for (int c = 0; c < 32; ++c) {
    float a0 = __shfl(g0, c, 32);
    float a1 = __shfl(g1, c, 32);
#pragma unroll
    for (int jj = 0; jj < 4; ++jj) {
      int j = lane * 4 + jj;
      acc[jj] += a0 * wS[c * 128 + j] + a1 * wS[(c + 32) * 128 + j];
    }
  }
#pragma unroll
  for (int jj = 0; jj < 4; ++jj) {
    int j = lane * 4 + jj;
    float val = pad ? 0.f : (acc[jj] + bS[j]);
    if (j < 64)
      K[((size_t)b * MKPAD + m) * 64 + j] = (_Float16)val;
    else
      Vt[((size_t)b * 64 + (j - 64)) * MKPAD + m] = (_Float16)val;
  }
}

// -------- Kernel 7: flash attention (WMMA f16) + fused output proj ----------
// Block: 128 threads (4 waves), each wave owns 16 query rows.
// LDS: K[736][64]f16 + Vt[64][736]f16 + Wt[64][64]f16 + 4x(16x64)f16 = 200 KB
// K+Vt staged by the Tensor Data Mover (one 1-row 8B-element descriptor:
// both buffers are contiguous in global ws AND in LDS).
__global__ __launch_bounds__(128) void attn_kernel(
    const _Float16* __restrict__ Q, const _Float16* __restrict__ K,
    const _Float16* __restrict__ Vt, const float* __restrict__ projW,
    const float* __restrict__ projB, float* __restrict__ out) {
  extern __shared__ char smem[];
  _Float16* Ksh = (_Float16*)smem;          // 736*64   (LDS offset 0)
  _Float16* Vsh = Ksh + MKPAD * 64;         // 64*736   (LDS offset 94208)
  _Float16* Wsh = Vsh + 64 * MKPAD;         // 64*64    (Wt[cout][cin])
  _Float16* Psh = Wsh + 64 * 64;            // 4 waves * 16*64

  const int b = blockIdx.y;
  const int qtile = blockIdx.x;
  const int tid = threadIdx.x;
  const int lane = tid & 31;
  const int wid = tid >> 5;
  const int half_id = lane >> 4;
  const int n16 = lane & 15;

  // --- TDM: one descriptor moves K (94,208B) + Vt (94,208B) = 23,552 x 8B ---
  if (wid == 0) {
    uint64_t ga = (uint64_t)(const void*)(K + (size_t)b * MKPAD * 64);
    uint32x4 g0;
    g0.x = 1u;                                   // count=1, user mode
    g0.y = 0u;                                   // lds_addr = 0 (Ksh)
    g0.z = (unsigned)(ga & 0xffffffffu);         // global_addr[31:0]
    g0.w = (unsigned)((ga >> 32) & 0x1ffffffu) | (2u << 30);  // [56:32], type=2
    int32x8 g1;
    g1[0] = (int)(3u << 16);                     // data_size = 8B
    g1[1] = (int)(0x5C00u << 16);                // tensor_dim0[15:0]=23552
    g1[2] = (int)(1u << 16);                     // tensor_dim0 hi=0, tensor_dim1=1
    g1[3] = (int)(0x5C00u << 16);                // tile_dim0 = 23552
    g1[4] = 1;                                   // tile_dim1 = 1
    g1[5] = 23552;                               // tensor_dim0_stride
    g1[6] = 0;
    g1[7] = 0;
    int32x4 gz4 = {0, 0, 0, 0};
    int32x8 gz8 = {0, 0, 0, 0, 0, 0, 0, 0};
    __builtin_amdgcn_tensor_load_to_lds(g0, g1, gz4, gz4, gz8, 0);
    __builtin_amdgcn_s_wait_tensorcnt(0);
  }
  // transpose projW -> f16 while the TDM runs
  for (int i = tid; i < 64 * 64; i += 128) {
    int cout = i >> 6, cin = i & 63;
    Wsh[i] = (_Float16)projW[cin * 64 + cout];
  }
  __syncthreads();

  // Q A-fragments (16x32 f16, contiguous k order): lane holds row M=lane&15,
// VGPR p holds K = 2p+16*half
  AFrag A0, A1;
  {
    int qrow = qtile * 64 + wid * 16 + n16;
    const _Float16* qp = Q + ((size_t)b * NQPAD + qrow) * 64;
#pragma unroll
    for (int p = 0; p < 8; ++p) {
      int k = 16 * half_id + 2 * p;
      A0.u[p] = *(const uint32_t*)(qp + k);
      A1.u[p] = *(const uint32_t*)(qp + 32 + k);
    }
  }
  AFrag ONE;  // all-ones f16 B fragment: row-sum accumulator trick
#pragma unroll
  for (int p = 0; p < 8; ++p) ONE.u[p] = 0x3C003C00u;

  float mrow[8];
  DFrag acc[5];  // [0..3] = O tiles, [4] = running row-sum tile (P x 1)
#pragma unroll
  for (int r = 0; r < 8; ++r) mrow[r] = -1e30f;
#pragma unroll
  for (int t = 0; t < 5; ++t)
#pragma unroll
    for (int r = 0; r < 8; ++r) acc[t].f[r] = 0.f;

  _Float16* Pme = Psh + wid * 16 * 64;  // per-wave scratch

  for (int kc = 0; kc < 23; ++kc) {  // 23 * 32 = 736 keys
    const int kbase = kc * 32;
    DFrag S[2];
#pragma unroll
    for (int j = 0; j < 2; ++j) {
      int key = kbase + 16 * j + n16;
      const _Float16* kp = Ksh + key * 64;
      AFrag B0, B1;  // B = K^T chunk (channels x keys)
#pragma unroll
      for (int p = 0; p < 8; ++p) {
        int c = half_id * 16 + 2 * p;
        B0.u[p] = *(const uint32_t*)(kp + c);
        B1.u[p] = *(const uint32_t*)(kp + 32 + c);
      }
      v8f z = {};
      z = __builtin_amdgcn_wmma_f32_16x16x32_f16(false, A0.v, false, B0.v,
                                                 (short)0, z, false, false);
      z = __builtin_amdgcn_wmma_f32_16x16x32_f16(false, A1.v, false, B1.v,
                                                 (short)0, z, false, false);
      S[j].v = z;
      if (key >= MK) {
#pragma unroll
        for (int r = 0; r < 8; ++r) S[j].f[r] = -1e30f;
      }
    }
    // 16-lane row-max all-reduce: pure-VALU DPP butterfly
    // (xor1=quad_perm(1,0,3,2), xor2=quad_perm(2,3,0,1),
    //  xor4≡row_half_mirror, xor8≡row_mirror once groups are uniform)
    float mx[8], s0v[8], s1v[8];
#pragma unroll
    for (int r = 0; r < 8; ++r) {
      s0v[r] = S[0].f[r];
      s1v[r] = S[1].f[r];
      mx[r] = fmaxf(s0v[r], s1v[r]);
    }
#pragma unroll
    for (int r = 0; r < 8; ++r) mx[r] = dpp_fmax_step<0xB1>(mx[r]);
#pragma unroll
    for (int r = 0; r < 8; ++r) mx[r] = dpp_fmax_step<0x4E>(mx[r]);
#pragma unroll
    for (int r = 0; r < 8; ++r) mx[r] = dpp_fmax_step<0x141>(mx[r]);
#pragma unroll
    for (int r = 0; r < 8; ++r) mx[r] = dpp_fmax_step<0x140>(mx[r]);
#pragma unroll
    for (int r = 0; r < 8; ++r) {
      float mnew = fmaxf(mrow[r], mx[r]);
      float alpha = __expf(mrow[r] - mnew);
      float p0 = __expf(s0v[r] - mnew);
      float p1 = __expf(s1v[r] - mnew);
      mrow[r] = mnew;
#pragma unroll
      for (int t = 0; t < 5; ++t) acc[t].f[r] *= alpha;
      int M = r + 8 * half_id;
      Pme[M * 32 + n16] = (_Float16)p0;
      Pme[M * 32 + 16 + n16] = (_Float16)p1;
    }
    // P: D-layout -> A-layout via per-wave LDS scratch
    AFrag PA;
#pragma unroll
    for (int p = 0; p < 8; ++p) {
      int k = 16 * half_id + 2 * p;
      PA.u[p] = *(const uint32_t*)(Pme + n16 * 32 + k);
    }
    // O += P * V  (B from transposed V: contiguous key pairs)
#pragma unroll
    for (int t = 0; t < 4; ++t) {
      AFrag BV;
      const _Float16* vp = Vsh + (16 * t + n16) * MKPAD + kbase + half_id * 16;
#pragma unroll
      for (int p = 0; p < 8; ++p) BV.u[p] = *(const uint32_t*)(vp + 2 * p);
      acc[t].v = __builtin_amdgcn_wmma_f32_16x16x32_f16(
          false, PA.v, false, BV.v, (short)0, acc[t].v, false, false);
    }
    // running softmax denominator: sum tile += P x ones
    acc[4].v = __builtin_amdgcn_wmma_f32_16x16x32_f16(
        false, PA.v, false, ONE.v, (short)0, acc[4].v, false, false);
  }

  // Normalize (denominator already broadcast in every lane of the sum tile)
  // and stage O as f16 for the fused projection
#pragma unroll
  for (int r = 0; r < 8; ++r) {
    float inv = 1.0f / acc[4].f[r];
    int M = r + 8 * half_id;
#pragma unroll
    for (int t = 0; t < 4; ++t)
      Pme[M * 64 + 16 * t + n16] = (_Float16)(acc[t].f[r] * inv);
  }
  AFrag AO0, AO1;
#pragma unroll
  for (int p = 0; p < 8; ++p) {
    int k = 16 * half_id + 2 * p;
    AO0.u[p] = *(const uint32_t*)(Pme + n16 * 64 + k);
    AO1.u[p] = *(const uint32_t*)(Pme + n16 * 64 + 32 + k);
  }
  DFrag outT[4];
#pragma unroll
  for (int t = 0; t < 4; ++t) {
    AFrag BW0, BW1;
    const _Float16* wp = Wsh + (16 * t + n16) * 64 + half_id * 16;
#pragma unroll
    for (int p = 0; p < 8; ++p) {
      BW0.u[p] = *(const uint32_t*)(wp + 2 * p);
      BW1.u[p] = *(const uint32_t*)(wp + 32 + 2 * p);
    }
    v8f z = {};
    z = __builtin_amdgcn_wmma_f32_16x16x32_f16(false, AO0.v, false, BW0.v,
                                               (short)0, z, false, false);
    z = __builtin_amdgcn_wmma_f32_16x16x32_f16(false, AO1.v, false, BW1.v,
                                               (short)0, z, false, false);
    outT[t].v = z;
  }
#pragma unroll
  for (int t = 0; t < 4; ++t) {
    float pb = projB[16 * t + n16];
#pragma unroll
    for (int r = 0; r < 8; ++r) {
      int q = qtile * 64 + wid * 16 + r + 8 * half_id;
      if (q < NQ)
        out[((size_t)b * NQ + q) * 64 + 16 * t + n16] = outT[t].f[r] + pb;
    }
  }
}

// ---------------------------------------------------------------------------
extern "C" void kernel_launch(void* const* d_in, const int* in_sizes, int n_in,
                              void* d_out, int out_size, void* d_ws,
                              size_t ws_size, hipStream_t stream) {
  const float* x      = (const float*)d_in[0];
  const float* Wq     = (const float*)d_in[1];
  const float* bq     = (const float*)d_in[2];
  const float* sr1_w  = (const float*)d_in[3];
  const float* sr1_b  = (const float*)d_in[4];
  const float* sr3_w  = (const float*)d_in[5];
  const float* sr3_b  = (const float*)d_in[6];
  const float* sr5_w  = (const float*)d_in[7];
  const float* sr5_b  = (const float*)d_in[8];
  const float* ln_g   = (const float*)d_in[9];
  const float* ln_b   = (const float*)d_in[10];
  const float* kv_w   = (const float*)d_in[11];
  const float* kv_b   = (const float*)d_in[12];
  const float* proj_w = (const float*)d_in[13];
  const float* proj_b = (const float*)d_in[14];
  float* out = (float*)d_out;

  char* ws = (char*)d_ws;
  _Float16* Q  = (_Float16*)(ws);
  float*    X1 = (float*)(ws + 12058624);
  _Float16* K  = (_Float16*)(ws + 15073280);
  _Float16* Vt = (_Float16*)(ws + 16580608);

  qproj_kernel<<<16 * 92, 256, 0, stream>>>(x, Wq, bq, Q);
  copy_zero_kernel<<<1572, 256, 0, stream>>>(x, X1);
  conv1_kernel<<<196, 256, 0, stream>>>(x, sr1_w, sr1_b, X1);
  conv3_kernel<<<392, 256, 0, stream>>>(x, sr3_w, sr3_b, X1);
  conv5_kernel<<<784, 256, 0, stream>>>(x, sr5_w, sr5_b, X1);
  lngelu_kv_kernel<<<1472, 256, 0, stream>>>(X1, kv_w, kv_b, ln_g, ln_b, K, Vt);

  // 200 KB dynamic LDS — fits CDNA5's 320 KB WGP LDS
  const size_t smem = (size_t)(MKPAD * 64 + 64 * MKPAD + 64 * 64 + 4 * 16 * 64) * 2;
  attn_kernel<<<dim3(92, 16), 128, smem, stream>>>(Q, K, Vt, proj_w, proj_b, out);
}
